// GroupedQueryAttention_57217554317356
// MI455X (gfx1250) — matrix operand order repaired and checked by reading, not compiled, not run on hardware
//
#include <hip/hip_runtime.h>

// ---------------------------------------------------------------------------
// GQA attention block for gfx1250 (MI455X). All matrix math runs on
// v_wmma_f32_16x16x32_f16 (wave32). f32 softmax/RoPE.
//   - GEMMs: B tile staged in LDS via GLOBAL_LOAD_ASYNC_TO_LDS (ASYNCcnt) +
//     global_prefetch of the next K-step
//   - Attention: 32 query rows / wave; K/V fragments reused by both halves;
//     causal mask only on diagonal tile; row-max via DPP ROW_XMASK (VALU),
//     row-sum via an extra WMMA against a ones matrix (no LDS round trips)
// ---------------------------------------------------------------------------

typedef _Float16 f16;
typedef __attribute__((ext_vector_type(16))) _Float16 v16h;
typedef __attribute__((ext_vector_type(8)))  _Float16 v8h;
typedef __attribute__((ext_vector_type(8)))  float    v8f;
typedef __attribute__((ext_vector_type(4)))  int      v4i;

// address-space-qualified v4i for the async-to-LDS builtin
typedef __attribute__((address_space(1))) v4i g_v4i;   // global
typedef __attribute__((address_space(3))) v4i l_v4i;   // LDS

#define BB    2
#define SS    2048
#define DD    2048
#define HH    32
#define KVH   8
#define HDIM  64
#define REPQ  4
#define SCALEF 0.125f   // 1/sqrt(64)

#if defined(__has_builtin)
#if __has_builtin(__builtin_amdgcn_global_load_async_to_lds_b128) && \
    __has_builtin(__builtin_amdgcn_s_wait_asynccnt)
#define HAVE_ASYNC_LDS 1
#endif
#if __has_builtin(__builtin_amdgcn_update_dpp)
#define HAVE_DPP 1
#endif
#endif
#ifndef HAVE_ASYNC_LDS
#define HAVE_ASYNC_LDS 0
#endif
#ifndef HAVE_DPP
#define HAVE_DPP 0
#endif

// ---------------- WMMA helpers -------------------------------------------------

__device__ __forceinline__ v8f wmma32(v16h a, v16h b, v8f c) {
  // D = A(16x32 f16) * B(32x16 f16) + C(16x16 f32)
  return __builtin_amdgcn_wmma_f32_16x16x32_f16(false, a, false, b, (short)0, c,
                                                false, false);
}

// A-fragment (16x32 f16), A row-major with leading dim `ld`, base at (m=0,k=0).
// Layout: lanes 0-15: M=lane, K0..15;
//         lanes 16-31: M=lane-16, K16..31.
__device__ __forceinline__ v16h load_a_frag(const f16* A, int ld) {
  int l = threadIdx.x & 31;
  const f16* row = A + (long)(l & 15) * ld + (l >> 4) * 16;
  union { v16h v; v8h h[2]; } u;
  u.h[0] = *(const v8h*)(row);        // K = sel*16 .. sel*16+7
  u.h[1] = *(const v8h*)(row + 8);    // K = sel*16+8 .. sel*16+15
  return u.v;
}

// B-fragment (32x16 f16) where element (k,n) = Bt[n*ld + k] (B stored
// transposed). Lanes 0-15: N=lane, K=0..15; lanes 16-31: K=16..31.
__device__ __forceinline__ v16h load_b_frag(const f16* Bt, int ld) {
  int l = threadIdx.x & 31;
  return *(const v16h*)(Bt + (long)(l & 15) * ld + (l >> 4) * 16);
}

// ---------------- half-wave (16-lane row) max reduction ------------------------

#if HAVE_DPP
// DPP16 ROW_XMASK (0x160|mask): lane i reads lane (i^mask) within its 16-lane
// row — pure VALU xor-butterfly, no LDS / DScnt involvement.
template <int CTRL>
__device__ __forceinline__ float dpp_mov_f32(float x) {
  int r = __builtin_amdgcn_update_dpp(0, __builtin_bit_cast(int, x),
                                      CTRL, 0xf, 0xf, true);
  return __builtin_bit_cast(float, r);
}
__device__ __forceinline__ float rowmax16(float x) {
  x = fmaxf(x, dpp_mov_f32<0x161>(x));   // xor 1
  x = fmaxf(x, dpp_mov_f32<0x162>(x));   // xor 2
  x = fmaxf(x, dpp_mov_f32<0x164>(x));   // xor 4
  x = fmaxf(x, dpp_mov_f32<0x168>(x));   // xor 8
  return x;
}
#else
__device__ __forceinline__ float rowmax16(float x) {
  x = fmaxf(x, __shfl_xor(x, 1, 32));
  x = fmaxf(x, __shfl_xor(x, 2, 32));
  x = fmaxf(x, __shfl_xor(x, 4, 32));
  x = fmaxf(x, __shfl_xor(x, 8, 32));
  return x;
}
#endif

// ---------------- elementwise staging kernels ---------------------------------

__global__ __launch_bounds__(256) void cast_f32_f16_kernel(
    const float* __restrict__ in, f16* __restrict__ out, long n) {
  long i = (long)blockIdx.x * blockDim.x + threadIdx.x;
  if (i < n) out[i] = (f16)in[i];
}

// in: [K,N] f32 row-major  ->  out: [N,K] f16 row-major ("Bt" for GEMM)
__global__ __launch_bounds__(256) void transpose_cast_kernel(
    const float* __restrict__ in, f16* __restrict__ out, int K, int N) {
  long i = (long)blockIdx.x * blockDim.x + threadIdx.x;
  if (i >= (long)K * N) return;
  int n = (int)(i % N);
  int k = (int)(i / N);
  out[(long)n * K + k] = (f16)in[i];
}

// RoPE + transpose to head-major: in [B,S,NH,HD] f16 -> out [B,NH,S,HD] f16
__global__ __launch_bounds__(256) void rope_transpose_kernel(
    const f16* __restrict__ in, const float* __restrict__ cosT,
    const float* __restrict__ sinT, f16* __restrict__ out, int NH) {
  long i = (long)blockIdx.x * blockDim.x + threadIdx.x;
  long total = (long)BB * SS * NH * HDIM;
  if (i >= total) return;
  int hd = (int)(i % HDIM);
  long t = i / HDIM;
  int h = (int)(t % NH);  t /= NH;
  int s = (int)(t % SS);
  int b = (int)(t / SS);
  long iin = (((long)b * SS + s) * NH + h) * HDIM + hd;
  float u  = (float)in[iin];
  float c  = cosT[s * HDIM + hd];
  float sn = sinT[s * HDIM + hd];
  float rot = (hd < HDIM / 2) ? -(float)in[iin + HDIM / 2]
                              :  (float)in[iin - HDIM / 2];
  out[(((long)b * NH + h) * SS + s) * HDIM + hd] = (f16)(u * c + rot * sn);
}

// V transpose: in [B,S,KV,HD] f16 -> out [B,KV,HD,S] f16
__global__ __launch_bounds__(256) void vtranspose_kernel(
    const f16* __restrict__ in, f16* __restrict__ out) {
  long i = (long)blockIdx.x * blockDim.x + threadIdx.x;
  long total = (long)BB * SS * KVH * HDIM;
  if (i >= total) return;
  int hd = (int)(i % HDIM);
  long t = i / HDIM;
  int kv = (int)(t % KVH);  t /= KVH;
  int s  = (int)(t % SS);
  int b  = (int)(t / SS);
  out[(((long)b * KVH + kv) * HDIM + hd) * SS + s] = in[i];
}

// ---------------- WMMA GEMM with LDS-staged B ---------------------------------
// C[M,N] = A[M,K](f16 row-major) * Bt[N,K](f16, B transposed).
// Block: 256 threads = 8 waves; block tile = 128 rows x 64 cols.
// The 64x32 B tile is staged once per K-step into LDS and shared by all waves.

template <typename OutT>
__global__ __launch_bounds__(256) void gemm_wmma_kernel(
    const f16* __restrict__ A, const f16* __restrict__ Bt,
    OutT* __restrict__ C, int M, int N, int K) {
  __shared__ f16 Bl[64][32];   // [n][k] 4 KB

  int t = threadIdx.x;
  int wave = t >> 5;
  int lane = t & 31;
  int m0 = (blockIdx.y * 8 + wave) * 16;   // grids are exact: m0 < M always
  int n0 = blockIdx.x * 64;

  // cooperative B-tile staging: 64 rows x 4 chunks of 8 f16 (16 B per thread)
  int bn = t >> 2;
  int bc = (t & 3) * 8;
  const f16* Bg = Bt + (long)(n0 + bn) * K + bc;

  const f16* Ab = A + (long)m0 * K;
  v8f acc[4] = {};

  for (int k = 0; k < K; k += 32) {
    __syncthreads();
#if HAVE_ASYNC_LDS
    // memory -> LDS DMA, no VGPR round-trip; tracked with ASYNCcnt
    __builtin_amdgcn_global_load_async_to_lds_b128(
        (g_v4i*)(Bg + k), (l_v4i*)(&Bl[bn][bc]), 0, 0);
    __builtin_prefetch(Bg + k + 32, 0, 3);   // global_prefetch_b8 (next K-step)
    __builtin_amdgcn_s_wait_asynccnt(0);     // own LDS writes complete
#else
    *(v8h*)(&Bl[bn][bc]) = *(const v8h*)(Bg + k);
    __builtin_prefetch(Bg + k + 32, 0, 3);
#endif
    __syncthreads();

    v16h a = load_a_frag(Ab + k, K);
#pragma unroll
    for (int f = 0; f < 4; ++f)
      acc[f] = wmma32(a, load_b_frag(&Bl[f * 16][0], 32), acc[f]);
  }

  int nn = lane & 15, mh = lane >> 4;  // C layout: row = r + 8*mh, col = nn
#pragma unroll
  for (int f = 0; f < 4; ++f)
#pragma unroll
    for (int r = 0; r < 8; ++r)
      C[(long)(m0 + r + 8 * mh) * N + n0 + f * 16 + nn] = (OutT)acc[f][r];
}

// ---------------- flash attention (causal, GQA) -------------------------------
// One wave per 32-query tile (two 16-row halves share K/V fragments);
// key tiles of 32; only the diagonal tile is masked. Qr [B,H,S,HD],
// Kr [B,KV,S,HD], Vt [B,KV,HD,S]; output AttO [B,S,H*HD] f16.

__global__ __launch_bounds__(256) void attn_kernel(
    const f16* __restrict__ Qr, const f16* __restrict__ Kr,
    const f16* __restrict__ Vt, f16* __restrict__ AttO) {
  __shared__ f16 Ptile[8][16 * 32];  // per-wave P staging (C->A re-layout)

  int wave = threadIdx.x >> 5;
  int lane = threadIdx.x & 31;
  int nn = lane & 15, mh = lane >> 4;

  int bh = blockIdx.y;
  int b = bh / HH, h = bh % HH;
  int kv = h / REPQ;
  int q0 = (blockIdx.x * 8 + wave) * 32;

  const f16* Qb = Qr + (((long)b * HH + h) * SS + q0) * HDIM;
  const f16* Kb = Kr + (((long)b * KVH + kv) * SS) * HDIM;
  const f16* Vb = Vt + (((long)b * KVH + kv) * HDIM) * SS;

  // q A-fragments: [rowhalf][khalf]
  v16h qa[2][2];
#pragma unroll
  for (int rh = 0; rh < 2; ++rh)
#pragma unroll
    for (int kh = 0; kh < 2; ++kh)
      qa[rh][kh] = load_a_frag(Qb + (long)rh * 16 * HDIM + kh * 32, HDIM);

  // ones B-matrix for row-sum WMMA (P x 1 -> row sums in every column)
  v16h ones;
#pragma unroll
  for (int i = 0; i < 16; ++i) ones[i] = (f16)1.0f;

  float mrow[2][8], lrow[2][8];
#pragma unroll
  for (int rh = 0; rh < 2; ++rh)
#pragma unroll
    for (int r = 0; r < 8; ++r) { mrow[rh][r] = -1e30f; lrow[rh][r] = 0.0f; }
  v8f acc[2][4] = {};

  f16* pw = &Ptile[wave][0];

  // process one 32-key tile; `masked` is a literal at each call site and folds
  auto tile = [&](int j, bool masked) {
    // K/V fragments for this tile, shared by both row halves
    v16h kb0 = load_b_frag(Kb + (long)j * HDIM, HDIM);
    v16h kb1 = load_b_frag(Kb + (long)j * HDIM + 32, HDIM);
    v16h kb2 = load_b_frag(Kb + (long)(j + 16) * HDIM, HDIM);
    v16h kb3 = load_b_frag(Kb + (long)(j + 16) * HDIM + 32, HDIM);
    v16h vb[4];
#pragma unroll
    for (int f = 0; f < 4; ++f)
      vb[f] = load_b_frag(Vb + (long)(f * 16) * SS + j, SS);

    // prefetch next tile's K/V (speculative; OOB prefetch is dropped)
    __builtin_prefetch(Kb + (long)(j + 32) * HDIM + (long)nn * HDIM, 0, 3);
    __builtin_prefetch(Vb + (long)nn * SS + j + 32, 0, 3);

#pragma unroll
    for (int rh = 0; rh < 2; ++rh) {
      v8f s0 = {}, s1 = {};
      s0 = wmma32(qa[rh][0], kb0, s0);
      s0 = wmma32(qa[rh][1], kb1, s0);
      s1 = wmma32(qa[rh][0], kb2, s1);
      s1 = wmma32(qa[rh][1], kb3, s1);

      float scv[8];
#pragma unroll
      for (int r = 0; r < 8; ++r) {
        float v0 = s0[r] * SCALEF;
        float v1 = s1[r] * SCALEF;
        if (masked) {  // diagonal tile only; selects, no EXEC change
          int qrow = q0 + rh * 16 + r + 8 * mh;
          v0 = (j + nn      > qrow) ? -1e30f : v0;
          v1 = (j + 16 + nn > qrow) ? -1e30f : v1;
        }

        float rmax = rowmax16(fmaxf(v0, v1));
        float mnew = fmaxf(mrow[rh][r], rmax);
        float sc = __expf(mrow[rh][r] - mnew);
        mrow[rh][r] = mnew;
        scv[r] = sc;

        float e0 = __expf(v0 - mnew);
        float e1 = __expf(v1 - mnew);
#pragma unroll
        for (int f = 0; f < 4; ++f) acc[rh][f][r] *= sc;

        // stage P (16x32) in LDS, row-major (same-wave DS ops are in-order)
        pw[(r + 8 * mh) * 32 + nn]      = (f16)e0;
        pw[(r + 8 * mh) * 32 + 16 + nn] = (f16)e1;
      }

      v16h pa = load_a_frag(pw, 32);  // P as f16 A-fragment (16x32)

      // row sums via WMMA: every column of sumf holds its row's sum of P
      v8f zero = {};
      v8f sumf = wmma32(pa, ones, zero);
#pragma unroll
      for (int r = 0; r < 8; ++r)
        lrow[rh][r] = lrow[rh][r] * scv[r] + sumf[r];

#pragma unroll
      for (int f = 0; f < 4; ++f)
        acc[rh][f] = wmma32(pa, vb[f], acc[rh][f]);
    }
  };

  // steady state: strictly-below-diagonal tiles need no masking
  for (int j = 0; j + 32 <= q0; j += 32) tile(j, false);
  // diagonal tile
  tile(q0, true);

#pragma unroll
  for (int rh = 0; rh < 2; ++rh)
#pragma unroll
    for (int r = 0; r < 8; ++r) {
      float inv = 1.0f / lrow[rh][r];
#pragma unroll
      for (int f = 0; f < 4; ++f) acc[rh][f][r] *= inv;
    }

  // AttO [B, S, H*HD]
#pragma unroll
  for (int rh = 0; rh < 2; ++rh)
#pragma unroll
    for (int f = 0; f < 4; ++f)
#pragma unroll
      for (int r = 0; r < 8; ++r) {
        long row = (long)b * SS + q0 + rh * 16 + r + 8 * mh;
        AttO[row * (HH * HDIM) + h * HDIM + f * 16 + nn] = (f16)acc[rh][f][r];
      }
}

// ---------------- host orchestration ------------------------------------------

extern "C" void kernel_launch(void* const* d_in, const int* in_sizes, int n_in,
                              void* d_out, int out_size, void* d_ws, size_t ws_size,
                              hipStream_t stream) {
  (void)in_sizes; (void)n_in; (void)out_size; (void)ws_size;

  const float* x    = (const float*)d_in[0];
  const float* cosT = (const float*)d_in[1];
  const float* sinT = (const float*)d_in[2];
  const float* Wq   = (const float*)d_in[3];
  const float* Wk   = (const float*)d_in[4];
  const float* Wv   = (const float*)d_in[5];
  const float* Wo   = (const float*)d_in[6];
  float* out = (float*)d_out;

  const long M  = (long)BB * SS;          // 4096
  const long NQ = (long)HH * HDIM;        // 2048
  const long NK = (long)KVH * HDIM;       // 512

  char* ws = (char*)d_ws;
  size_t off = 0;
  auto take = [&](long elems) -> f16* {
    f16* p = (f16*)(ws + off);
    off += (((size_t)elems * sizeof(f16)) + 255) & ~(size_t)255;
    return p;
  };

  f16* xh  = take(M * DD);        // x in f16 (reused for AttO later)
  f16* Wqt = take(NQ * DD);       // [N,K]
  f16* Wkt = take(NK * DD);
  f16* Wvt = take(NK * DD);
  f16* Wot = take(DD * NQ);       // [N=D, K=H*HD]
  f16* Qh  = take(M * NQ);        // [B,S,H,HD]  pre-RoPE
  f16* Kh  = take(M * NK);        // [B,S,KV,HD] pre-RoPE
  f16* Vh  = take(M * NK);        // [B,S,KV,HD]
  f16* Qr  = take(M * NQ);        // [B,H,S,HD]
  f16* Kr  = take(M * NK);        // [B,KV,S,HD]
  f16* Vt  = take(M * NK);        // [B,KV,HD,S]
  f16* AttO = xh;                 // xh is dead after the QKV GEMMs

  const int T = 256;
  auto blocks = [](long n, int t) { return (unsigned)((n + t - 1) / t); };

  // 1) cast x
  cast_f32_f16_kernel<<<blocks(M * DD, T), T, 0, stream>>>(x, xh, M * DD);

  // 2) transpose+cast weights to [N,K] f16
  transpose_cast_kernel<<<blocks((long)DD * NQ, T), T, 0, stream>>>(Wq, Wqt, DD, (int)NQ);
  transpose_cast_kernel<<<blocks((long)DD * NK, T), T, 0, stream>>>(Wk, Wkt, DD, (int)NK);
  transpose_cast_kernel<<<blocks((long)DD * NK, T), T, 0, stream>>>(Wv, Wvt, DD, (int)NK);
  transpose_cast_kernel<<<blocks((long)NQ * DD, T), T, 0, stream>>>(Wo, Wot, (int)NQ, DD);

  // 3) QKV projection GEMMs (f16 out)
  {
    dim3 g((unsigned)(NQ / 64), (unsigned)(M / 16 / 8));
    gemm_wmma_kernel<f16><<<g, T, 0, stream>>>(xh, Wqt, Qh, (int)M, (int)NQ, DD);
  }
  {
    dim3 g((unsigned)(NK / 64), (unsigned)(M / 16 / 8));
    gemm_wmma_kernel<f16><<<g, T, 0, stream>>>(xh, Wkt, Kh, (int)M, (int)NK, DD);
    gemm_wmma_kernel<f16><<<g, T, 0, stream>>>(xh, Wvt, Vh, (int)M, (int)NK, DD);
  }

  // 4) RoPE + head-major transposes
  rope_transpose_kernel<<<blocks(M * NQ, T), T, 0, stream>>>(Qh, cosT, sinT, Qr, HH);
  rope_transpose_kernel<<<blocks(M * NK, T), T, 0, stream>>>(Kh, cosT, sinT, Kr, KVH);
  vtranspose_kernel<<<blocks(M * NK, T), T, 0, stream>>>(Vh, Vt);

  // 5) causal flash attention (writes AttO over xh's region)
  {
    dim3 g((unsigned)(SS / (32 * 8)), (unsigned)(BB * HH));
    attn_kernel<<<g, T, 0, stream>>>(Qr, Kr, Vt, AttO);
  }

  // 6) output projection (f32 out)
  {
    dim3 g((unsigned)(DD / 64), (unsigned)(M / 16 / 8));
    gemm_wmma_kernel<float><<<g, T, 0, stream>>>(AttO, Wot, out, (int)M, DD, (int)NQ);
  }
}
